// SparseConvLSTM_26285199851965
// MI455X (gfx1250) — compile-verified, matrix operand order checked by reading, not run on hardware
//
#include <hip/hip_runtime.h>
#include <hip/hip_bf16.h>

// CDNA5 (gfx1250) wave32 WMMA types
typedef __bf16 bf16;
typedef __attribute__((ext_vector_type(16))) __bf16 v16bf;
typedef __attribute__((ext_vector_type(8)))  float  v8f;

#define T_STEPS 8
#define NVOX    50000
#define CIN     64
#define HDIM    64
#define K3      27
#define KCH     128   // conv input channels (CIN+H == 2H == 128)
#define NOUT    256   // 4*H gate channels
#define M_TILE  32    // voxel rows per block
#define LDA     144   // ldsA stride (bf16 elems): 288B = 9*32 -> keeps v16bf loads 32B-aligned
#define LDZ     272   // ldsZ stride (f32), padded for bank spread
#define WFRAG_PER_K 32768  // 16 ntiles * 4 ksteps * 32 lanes * 16 halves = 128*256

// Pack f32 weights into bf16 WMMA-B fragments:
// Wb[(((k*16 + nt)*4 + ks)*32 + lane)*16 + h] <- W[k][kc][n]
//   n  = nt*16 + (lane & 15)
//   kc = ks*32 + (lane>>4)*16 + h
__global__ void prep_weights(const float* __restrict__ W, bf16* __restrict__ Wb, int total) {
  int o = blockIdx.x * blockDim.x + threadIdx.x;
  if (o >= total) return;
  int h2   = o & 15;
  int lane = (o >> 4) & 31;
  int ks   = (o >> 9) & 3;
  int nt   = (o >> 11) & 15;
  int k    = o >> 15;
  int n  = nt * 16 + (lane & 15);
  int kc = ks * 32 + ((lane >> 4) * 16) + h2;
  Wb[o] = (bf16)W[(size_t)k * (KCH * NOUT) + (size_t)kc * NOUT + n];
}

__device__ __forceinline__ float sigmoidf(float x) { return 1.0f / (1.0f + __expf(-x)); }

// One conv+LSTM step for one layer: z = sum_k gather(concat(srcA,srcH))[nbr]@W[k] + b,
// then c' = sig(f)*c + sig(i)*tanh(g);  h' = sig(o)*tanh(c').
__global__ __launch_bounds__(256)
void subm_conv_lstm_step(const float* __restrict__ srcA,   // [N,64] low channels (x_t or h of layer 0)
                         const float* __restrict__ srcH,   // [N,64] high channels (own h_prev)
                         const int*   __restrict__ nbr,    // [N,27], sentinel == NVOX
                         const bf16*  __restrict__ Wb,     // packed bf16 fragments
                         const float* __restrict__ bias,   // [256]
                         float* __restrict__ cbuf,         // [N,64] cell state, updated in place
                         float* __restrict__ hout,         // [N,64] h_next
                         float* __restrict__ hs_out)       // optional extra h output (d_out slice)
{
  __shared__ __align__(32) bf16  ldsA[M_TILE * LDA];
  __shared__ __align__(16) float ldsZ[M_TILE * LDZ];

  const int tid  = threadIdx.x;
  const int lane = tid & 31;
  const int wave = tid >> 5;
  const int blockRow = blockIdx.x * M_TILE;

  const int mrow   = (wave & 1) * 16;     // wave's M sub-tile: rows 0..15 or 16..31
  const int n_wave = (wave >> 1) * 64;    // wave's 64-column slab of the 256 gate cols

  v8f acc[4] = {};                        // 16x64 f32 accumulators (4 x 16x16 tiles)

  const int gr = tid >> 3;                // gather row 0..31
  const int gq = tid & 7;                 // gather 16-channel chunk 0..7

  for (int k = 0; k < K3; ++k) {
    if (k) __syncthreads();               // protect ldsA readers of previous offset

    // ---- gather A tile: 32 rows x 128ch of concat(srcA,srcH) at neighbor idx, f32->bf16 ----
    {
      int g   = blockRow + gr;
      int idx = NVOX;                                   // sentinel -> zero row
      if (g < NVOX) idx = nbr[g * K3 + k];
      bf16* dst = &ldsA[gr * LDA + gq * 16];
      if (idx < NVOX) {
        const float4* src = (gq < 4)
            ? (const float4*)(srcA + (size_t)idx * HDIM + gq * 16)
            : (const float4*)(srcH + (size_t)idx * HDIM + (gq - 4) * 16);
        #pragma unroll
        for (int v = 0; v < 4; ++v) {
          float4 f = src[v];
          dst[v * 4 + 0] = (bf16)f.x; dst[v * 4 + 1] = (bf16)f.y;
          dst[v * 4 + 2] = (bf16)f.z; dst[v * 4 + 3] = (bf16)f.w;
        }
      } else {
        #pragma unroll
        for (int v = 0; v < 16; ++v) dst[v] = (bf16)0.0f;
      }
    }
    // L2 prefetch of next offset's weight fragments (global_prefetch_b8)
    if (k + 1 < K3)
      __builtin_prefetch(Wb + (size_t)(k + 1) * WFRAG_PER_K + tid * 128, 0, 1);
    __syncthreads();

    // ---- WMMA: acc += A(16x32 bf16) x W[k](32x16 bf16), f32 accumulate ----
    const bf16* wk = Wb + (size_t)k * WFRAG_PER_K;
    #pragma unroll
    for (int ks = 0; ks < 4; ++ks) {
      const int arow = mrow + (lane & 15);
      const int koff = ks * 32 + ((lane >> 4) * 16);
      v16bf afrag = *(const v16bf*)&ldsA[arow * LDA + koff];
      #pragma unroll
      for (int i = 0; i < 4; ++i) {
        const int nt = (n_wave >> 4) + i;
        v16bf bfrag = *(const v16bf*)(wk + (((size_t)nt * 4 + ks) * 32 + lane) * 16);
        acc[i] = __builtin_amdgcn_wmma_f32_16x16x32_bf16(
            /*neg_a=*/false, afrag, /*neg_b=*/false, bfrag,
            /*c_mod=*/(short)0, acc[i], /*reuse_a=*/false, /*reuse_b=*/false);
      }
    }
  }

  // ---- spill gate pre-activations to LDS (C-fragment layout: vgpr r, lanes>=16 -> M+8) ----
  #pragma unroll
  for (int i = 0; i < 4; ++i) {
    const int col   = n_wave + i * 16 + (lane & 15);
    const int rbase = mrow + ((lane >> 4) * 8);
    #pragma unroll
    for (int r = 0; r < 8; ++r)
      ldsZ[(rbase + r) * LDZ + col] = acc[i][r];
  }
  __syncthreads();

  // ---- fused LSTM gating on the 32x64 tile ----
  const int j  = tid & 63;   // hidden channel
  const int r0 = tid >> 6;   // 0..3
  #pragma unroll
  for (int s = 0; s < 8; ++s) {
    const int r = r0 + s * 4;           // 0..31
    const int g = blockRow + r;
    if (g < NVOX) {
      float zi = ldsZ[r * LDZ +       j] + bias[      j];
      float zf = ldsZ[r * LDZ +  64 + j] + bias[ 64 + j];
      float zo = ldsZ[r * LDZ + 128 + j] + bias[128 + j];
      float zg = ldsZ[r * LDZ + 192 + j] + bias[192 + j];
      float cold = cbuf[(size_t)g * HDIM + j];
      float cn = sigmoidf(zf) * cold + sigmoidf(zi) * tanhf(zg);
      float hn = sigmoidf(zo) * tanhf(cn);
      cbuf[(size_t)g * HDIM + j] = cn;
      hout[(size_t)g * HDIM + j] = hn;
      if (hs_out) hs_out[(size_t)g * HDIM + j] = hn;
    }
  }
}

extern "C" void kernel_launch(void* const* d_in, const int* in_sizes, int n_in,
                              void* d_out, int out_size, void* d_ws, size_t ws_size,
                              hipStream_t stream) {
  (void)in_sizes; (void)n_in; (void)out_size; (void)ws_size;
  const float* x   = (const float*)d_in[0];   // [T,N,64]
  const float* W0  = (const float*)d_in[1];   // [27,128,256]
  const float* b0  = (const float*)d_in[2];   // [256]
  const float* W1  = (const float*)d_in[3];   // [27,128,256]
  const float* b1  = (const float*)d_in[4];   // [256]
  const int*   nbr = (const int*)d_in[5];     // [N,27]
  float* out = (float*)d_out;                 // [T,N,64]

  // workspace carve-out (256B aligned)
  char* ws = (char*)d_ws;
  size_t off = 0;
  auto grab = [&](size_t bytes) -> void* {
    off = (off + 255) & ~(size_t)255;
    void* p = ws + off;
    off += bytes;
    return p;
  };
  const size_t wfrag  = (size_t)K3 * WFRAG_PER_K;            // bf16 elems per layer
  bf16*  Wb0 = (bf16*)grab(wfrag * sizeof(bf16));
  bf16*  Wb1 = (bf16*)grab(wfrag * sizeof(bf16));
  const size_t hbytes = (size_t)NVOX * HDIM * sizeof(float);
  float* h0a = (float*)grab(hbytes);
  float* h0b = (float*)grab(hbytes);
  float* h1a = (float*)grab(hbytes);
  float* h1b = (float*)grab(hbytes);
  float* c0  = (float*)grab(hbytes);
  float* c1  = (float*)grab(hbytes);

  // deterministic init each call (graph-capture-safe memsets)
  hipMemsetAsync(h0a, 0, hbytes, stream);
  hipMemsetAsync(h1a, 0, hbytes, stream);
  hipMemsetAsync(c0,  0, hbytes, stream);
  hipMemsetAsync(c1,  0, hbytes, stream);

  {
    int total = (int)wfrag;
    int blk = 256, grd = (total + blk - 1) / blk;
    prep_weights<<<grd, blk, 0, stream>>>(W0, Wb0, total);
    prep_weights<<<grd, blk, 0, stream>>>(W1, Wb1, total);
  }

  const int grd = (NVOX + M_TILE - 1) / M_TILE;   // 1563 blocks x 8 wave32
  for (int t = 0; t < T_STEPS; ++t) {
    float* h0c = (t & 1) ? h0b : h0a;  float* h0n = (t & 1) ? h0a : h0b;
    float* h1c = (t & 1) ? h1b : h1a;  float* h1n = (t & 1) ? h1a : h1b;
    // layer 0: input = concat(x_t, h0_prev)
    subm_conv_lstm_step<<<grd, 256, 0, stream>>>(
        x + (size_t)t * NVOX * CIN, h0c, nbr, Wb0, b0, c0, h0n, nullptr);
    // layer 1: input = concat(h0_t, h1_prev); h also streamed to d_out[t]
    subm_conv_lstm_step<<<grd, 256, 0, stream>>>(
        h0n, h1c, nbr, Wb1, b1, c1, h1n, out + (size_t)t * NVOX * HDIM);
  }
}
